// GCN_498216206706
// MI455X (gfx1250) — matrix operand order repaired and checked by reading, not compiled, not run on hardware
//
#include <hip/hip_runtime.h>

typedef __attribute__((ext_vector_type(16))) __bf16 v16bf;
typedef __attribute__((ext_vector_type(8)))  float  v8f;

#define GCN_IN_C 128
#define GCN_HID  64
#define GCN_NCLS 40

// ---------------------------------------------------------------- utilities
__global__ void gcn_zero_kernel(float* __restrict__ p, long long n) {
  long long i = (long long)blockIdx.x * blockDim.x + threadIdx.x;
  if (i < n) p[i] = 0.0f;
}

// deg[dst[e]] += 1 over the E real edges (self-loops folded in later as +1)
__global__ void gcn_deg_kernel(float* __restrict__ deg,
                               const long long* __restrict__ dst, long long E) {
  long long e = (long long)blockIdx.x * blockDim.x + threadIdx.x;
  if (e < E) atomicAdd(&deg[dst[e]], 1.0f);
}

// dis[i] = rsqrt(deg[i] + 1)   (deg+1 >= 1 always, matches reference gcn_norm)
__global__ void gcn_dis_kernel(float* __restrict__ deg_dis, long long n) {
  long long i = (long long)blockIdx.x * blockDim.x + threadIdx.x;
  if (i < n) deg_dis[i] = rsqrtf(deg_dis[i] + 1.0f);
}

__global__ void gcn_norm_kernel(float* __restrict__ norm,
                                const long long* __restrict__ src,
                                const long long* __restrict__ dst,
                                const float* __restrict__ dis, long long E) {
  long long e = (long long)blockIdx.x * blockDim.x + threadIdx.x;
  if (e < E) norm[e] = dis[src[e]] * dis[dst[e]];
}

// ---------------------------------------------------------------- WMMA GEMM
// C[M,NC] = A[M,K] @ B[K,NC], fp32 in/out, computed with bf16 hi/lo split
// (3x v_wmma_f32_16x16x32_bf16 per 32-wide K chunk => ~fp32 accuracy).
// One wave owns one 16x16 output tile. K, NC are compile-time so all B-load
// strides fold into immediate offsets; K must be a multiple of 32.
// Out-of-range B columns are CLAMPED (not predicated): WMMA column n only
// feeds D column n, whose store is masked for exactly those lanes.
template <int K, int NC>
__global__ void gcn_wmma_gemm_kernel(const float* __restrict__ A,
                                     const float* __restrict__ B,
                                     float* __restrict__ C,
                                     int M, int ntn, int ntiles) {
  const int wave = blockIdx.x * (blockDim.x >> 5) + (threadIdx.x >> 5);
  if (wave >= ntiles) return;               // wave-uniform exit: EXEC stays all-1s
  const int lane = threadIdx.x & 31;
  const int l15  = lane & 15;
  const int hi   = lane >> 4;               // lane-half selector per ISA layouts
  const int mt   = wave / ntn;
  const int nt   = wave - mt * ntn;

  int rowA = mt * 16 + l15;
  if (rowA >= M) rowA = M - 1;              // clamp (store is masked below)
  const int  colB  = nt * 16 + l15;
  const bool colOK = colB < NC;
  const int  colC  = colOK ? colB : (NC - 1);   // clamped load column

  v8f acc = {0.f, 0.f, 0.f, 0.f, 0.f, 0.f, 0.f, 0.f};

  const float* arowBase = A + (size_t)rowA * K + (hi << 4);
  const float* bcolBase = B + (size_t)(hi << 4) * NC + colC;

#pragma unroll
  for (int kc = 0; kc < K; kc += 32) {
    // A tile 16x32: lane(0-15)=row M, element e -> K = e + hi*16
    const float* arow = arowBase + kc;
    float a[16];
#pragma unroll
    for (int e2 = 0; e2 < 8; ++e2) {        // contiguous K pairs -> b64/b128 loads
      const int koff = e2 << 1;
      const float2 v = *reinterpret_cast<const float2*>(arow + koff);
      a[2 * e2]     = v.x;
      a[2 * e2 + 1] = v.y;
    }
    // B tile 32x16: lane(0-15)=col N, element e -> K = kc + e + hi*16
    const float* bcol = bcolBase + (size_t)kc * NC;
    float b[16];
#pragma unroll
    for (int e = 0; e < 16; ++e) {
      b[e] = bcol[e * NC];                  // e*NC is an immediate offset
    }
    // bf16 hi/lo split
    v16bf ah, al, bh, bl;
#pragma unroll
    for (int e = 0; e < 16; ++e) {
      const __bf16 xh = (__bf16)a[e];
      ah[e] = xh;
      al[e] = (__bf16)(a[e] - (float)xh);
      const __bf16 yh = (__bf16)b[e];
      bh[e] = yh;
      bl[e] = (__bf16)(b[e] - (float)yh);
    }
    acc = __builtin_amdgcn_wmma_f32_16x16x32_bf16(false, ah, false, bh, (short)0, acc, false, false);
    acc = __builtin_amdgcn_wmma_f32_16x16x32_bf16(false, al, false, bh, (short)0, acc, false, false);
    acc = __builtin_amdgcn_wmma_f32_16x16x32_bf16(false, ah, false, bl, (short)0, acc, false, false);
  }

  // C/D layout: element r -> M = r + hi*8, N = lane&15
  if (colOK) {
#pragma unroll
    for (int r = 0; r < 8; ++r) {
      const int m = mt * 16 + r + (hi << 3);
      if (m < M) C[(size_t)m * NC + colB] = acc[r];
    }
  }
}

// ---------------------------------------------------------------- edge scatter
// agg[dst[e], c] += h[src[e], c] * norm[e]; 64 lanes per edge (c>=C lanes idle)
__global__ void gcn_scatter_kernel(const float* __restrict__ h,
                                   float* __restrict__ agg,
                                   const long long* __restrict__ src,
                                   const long long* __restrict__ dst,
                                   const float* __restrict__ norm,
                                   long long E, int C) {
  const long long idx = (long long)blockIdx.x * blockDim.x + threadIdx.x;
  const long long e = idx >> 6;
  const int c = (int)(idx & 63);
  if (e >= E || c >= C) return;
  const long long s = src[e];
  const long long d = dst[e];
  atomicAdd(&agg[d * C + c], h[s * C + c] * norm[e]);
}

// out[i,c] = act( agg[i,c] + h[i,c]*dis[i]^2 + bias[c] )  (self-loop folded in)
__global__ void gcn_epilogue_kernel(const float* __restrict__ agg,
                                    const float* __restrict__ hself,
                                    const float* __restrict__ dis,
                                    const float* __restrict__ bias,
                                    float* __restrict__ out,
                                    long long Nn, int C, int do_relu) {
  const long long idx = (long long)blockIdx.x * blockDim.x + threadIdx.x;
  if (idx >= Nn * (long long)C) return;
  const long long i = idx / C;
  const int c = (int)(idx - i * C);
  const float d = dis[i];
  float v = agg[idx] + hself[idx] * d * d + bias[c];
  out[idx] = do_relu ? fmaxf(v, 0.0f) : v;
}

// ---------------------------------------------------------------- launcher
extern "C" void kernel_launch(void* const* d_in, const int* in_sizes, int n_in,
                              void* d_out, int out_size, void* d_ws, size_t ws_size,
                              hipStream_t stream) {
  (void)n_in; (void)out_size; (void)ws_size;
  const float*     x  = (const float*)d_in[0];
  const long long* ei = (const long long*)d_in[1];   // int64 per reference
  const float*     W1 = (const float*)d_in[2];
  const float*     b1 = (const float*)d_in[3];
  const float*     W2 = (const float*)d_in[4];
  const float*     b2 = (const float*)d_in[5];

  const long long N = in_sizes[0] / GCN_IN_C;
  const long long E = in_sizes[1] / 2;
  const long long* src = ei;
  const long long* dst = ei + E;

  // workspace layout (floats, 256B aligned regions)
  float* ws = (float*)d_ws;
  size_t o = 0;
  float* dis  = ws + o; o += (size_t)((N + 63) & ~63LL);
  float* norm = ws + o; o += (size_t)((E + 63) & ~63LL);
  float* h1   = ws + o; o += (size_t)N * GCN_HID;
  float* agg1 = ws + o; o += (size_t)N * GCN_HID;
  float* h2   = h1;    // reuse: h1 dead after epilogue1
  float* agg2 = agg1;  // reuse: z1 (==agg1) dead after GEMM2 reads it

  const int T = 256;
  auto blocks = [](long long n) { return (unsigned)((n + 255) / 256); };

  // --- normalization ---
  gcn_zero_kernel<<<blocks(N), T, 0, stream>>>(dis, N);
  gcn_deg_kernel<<<blocks(E), T, 0, stream>>>(dis, dst, E);
  gcn_dis_kernel<<<blocks(N), T, 0, stream>>>(dis, N);
  gcn_norm_kernel<<<blocks(E), T, 0, stream>>>(norm, src, dst, dis, E);

  // --- layer 1: h1 = x @ W1 ---
  {
    const int ntm = (int)((N + 15) / 16);
    const int ntn = GCN_HID / 16;
    const int ntiles = ntm * ntn;
    gcn_wmma_gemm_kernel<GCN_IN_C, GCN_HID><<<(ntiles + 7) / 8, T, 0, stream>>>(
        x, W1, h1, (int)N, ntn, ntiles);
  }
  gcn_zero_kernel<<<blocks(N * GCN_HID), T, 0, stream>>>(agg1, N * GCN_HID);
  gcn_scatter_kernel<<<blocks(E * 64), T, 0, stream>>>(h1, agg1, src, dst, norm, E, GCN_HID);
  gcn_epilogue_kernel<<<blocks(N * GCN_HID), T, 0, stream>>>(
      agg1, h1, dis, b1, agg1, N, GCN_HID, 1);   // z1 = relu(...), in place

  // --- layer 2: h2 = z1 @ W2 ---
  {
    const int ntm = (int)((N + 15) / 16);
    const int ntn = (GCN_NCLS + 15) / 16;
    const int ntiles = ntm * ntn;
    gcn_wmma_gemm_kernel<GCN_HID, GCN_NCLS><<<(ntiles + 7) / 8, T, 0, stream>>>(
        agg1, W2, h2, (int)N, ntn, ntiles);
  }
  gcn_zero_kernel<<<blocks(N * GCN_NCLS), T, 0, stream>>>(agg2, N * GCN_NCLS);
  gcn_scatter_kernel<<<blocks(E * 64), T, 0, stream>>>(h2, agg2, src, dst, norm, E, GCN_NCLS);
  gcn_epilogue_kernel<<<blocks(N * GCN_NCLS), T, 0, stream>>>(
      agg2, h2, dis, b2, (float*)d_out, N, GCN_NCLS, 0);
}
